// MutiGraph_47622597378685
// MI455X (gfx1250) — matrix operand order repaired and checked by reading, not compiled, not run on hardware
//
#include <hip/hip_runtime.h>
#include <hip/hip_bf16.h>
#include <math.h>

#define DD 300
#define HH 4
#define KP 320          // K padded to multiple of 32
#define GCOLS 1200      // H*D
#define GPAD 1216       // GCOLS padded to multiple of 64 (4 N-tiles per wave)
#define SPAD 320        // sage cols padded to multiple of 64

typedef __attribute__((ext_vector_type(8)))  __bf16 bf16x8;
typedef __attribute__((ext_vector_type(16))) __bf16 bf16x16;
typedef __attribute__((ext_vector_type(8)))  float  v8f;

__device__ __forceinline__ unsigned short f32_to_bf16(float f) {
    unsigned int x = __float_as_uint(f);
    unsigned int r = x + 0x7FFFu + ((x >> 16) & 1u);   // round-to-nearest-even
    return (unsigned short)(r >> 16);
}

__device__ __forceinline__ void atomicMaxF(float* addr, float v) {
    if (v >= 0.0f) atomicMax((int*)addr, __float_as_int(v));
    else           atomicMin((unsigned int*)addr, __float_as_uint(v));
}

// ---------------- utility fills / conversions ----------------

__global__ void fill_f32(float* p, float v, int n) {
    int i = blockIdx.x * blockDim.x + threadIdx.x;
    if (i < n) p[i] = v;
}

// src [rows][cols] f32 row-major -> dst [rows][ldd] bf16, zero-padded cols
__global__ void conv_rows_bf16(const float* __restrict__ src, unsigned short* __restrict__ dst,
                               int rows, int cols, int ldd) {
    int i = blockIdx.x * blockDim.x + threadIdx.x;
    int total = rows * ldd;
    if (i >= total) return;
    int r = i / ldd, j = i - r * ldd;
    float v = (j < cols) ? src[(size_t)r * cols + j] : 0.0f;
    dst[i] = f32_to_bf16(v);
}

// W [k][ncols] f32 -> Bt [npad][KP] bf16 (column-major / transposed, zero pad)
__global__ void conv_bt_bf16(const float* __restrict__ w, unsigned short* __restrict__ bt,
                             int k, int ncols, int npad) {
    int i = blockIdx.x * blockDim.x + threadIdx.x;
    int total = npad * KP;
    if (i >= total) return;
    int n = i / KP, kk = i - n * KP;
    float v = (kk < k && n < ncols) ? w[(size_t)kk * ncols + n] : 0.0f;
    bt[i] = f32_to_bf16(v);
}

// ---------------- WMMA bf16 GEMM, 32x64 register-blocked per wave ----------------
// C[M][ldc] = A[M][KP] * Bt[nBlocks*64][KP]^T

__device__ __forceinline__ bf16x16 ld_a16(const unsigned short* p) {
    bf16x8 lo = *(const bf16x8*)(p);
    bf16x8 hi = *(const bf16x8*)(p + 16);   // A layout: K in {koff..+7, 16+koff..+7}
    return __builtin_shufflevector(lo, hi, 0,1,2,3,4,5,6,7,8,9,10,11,12,13,14,15);
}
__device__ __forceinline__ bf16x16 ld_b16(const unsigned short* p) {
    bf16x8 lo = *(const bf16x8*)(p);
    bf16x8 hi = *(const bf16x8*)(p + 16);   // B layout: K in {koff..+7, 16+koff..+7}
    return __builtin_shufflevector(lo, hi, 0,1,2,3,4,5,6,7,8,9,10,11,12,13,14,15);
}

__global__ void wmma_gemm_bf16_blk(const unsigned short* __restrict__ A,
                                   const unsigned short* __restrict__ Bt,
                                   float* __restrict__ C,
                                   int M, int mBlocks, int nBlocks, int ncReal, int ldc) {
    int wid = (blockIdx.x * blockDim.x + threadIdx.x) >> 5;
    if (wid >= mBlocks * nBlocks) return;
    int bm = wid / nBlocks;
    int bn = wid - bm * nBlocks;
    int lane = threadIdx.x & 31;
    int half = lane >> 4;          // 0 or 1
    int l16  = lane & 15;

    int ar0 = bm * 32 + l16;       if (ar0 >= M) ar0 = M - 1;
    int ar1 = bm * 32 + 16 + l16;  if (ar1 >= M) ar1 = M - 1;
    const unsigned short* ap0 = A + (size_t)ar0 * KP + half * 8;
    const unsigned short* ap1 = A + (size_t)ar1 * KP + half * 8;
    const unsigned short* bp  = Bt + (size_t)(bn * 64 + l16) * KP + half * 8;

    v8f acc[2][4] = {};
#pragma unroll
    for (int k0 = 0; k0 < KP; k0 += 32) {
        bf16x16 a0 = ld_a16(ap0 + k0);
        bf16x16 a1 = ld_a16(ap1 + k0);
        bf16x16 b0 = ld_b16(bp + k0);
        bf16x16 b1 = ld_b16(bp + k0 + (size_t)16 * KP);
        bf16x16 b2 = ld_b16(bp + k0 + (size_t)32 * KP);
        bf16x16 b3 = ld_b16(bp + k0 + (size_t)48 * KP);
        acc[0][0] = __builtin_amdgcn_wmma_f32_16x16x32_bf16(false, a0, false, b0, (short)0, acc[0][0], false, false);
        acc[0][1] = __builtin_amdgcn_wmma_f32_16x16x32_bf16(false, a0, false, b1, (short)0, acc[0][1], false, false);
        acc[0][2] = __builtin_amdgcn_wmma_f32_16x16x32_bf16(false, a0, false, b2, (short)0, acc[0][2], false, false);
        acc[0][3] = __builtin_amdgcn_wmma_f32_16x16x32_bf16(false, a0, false, b3, (short)0, acc[0][3], false, false);
        acc[1][0] = __builtin_amdgcn_wmma_f32_16x16x32_bf16(false, a1, false, b0, (short)0, acc[1][0], false, false);
        acc[1][1] = __builtin_amdgcn_wmma_f32_16x16x32_bf16(false, a1, false, b1, (short)0, acc[1][1], false, false);
        acc[1][2] = __builtin_amdgcn_wmma_f32_16x16x32_bf16(false, a1, false, b2, (short)0, acc[1][2], false, false);
        acc[1][3] = __builtin_amdgcn_wmma_f32_16x16x32_bf16(false, a1, false, b3, (short)0, acc[1][3], false, false);
    }

#pragma unroll
    for (int i = 0; i < 2; ++i) {
        int crow0 = bm * 32 + i * 16 + half * 8;
#pragma unroll
        for (int j = 0; j < 4; ++j) {
            int ccol = bn * 64 + j * 16 + l16;
            if (ccol < ncReal) {
#pragma unroll
                for (int r = 0; r < 8; ++r) {
                    int cr = crow0 + r;
                    if (cr < M) C[(size_t)cr * ldc + ccol] = acc[i][j][r];
                }
            }
        }
    }
}

// ---------------- GAT attention logits ----------------

// one wave per (n,h): el/er dot products over D
__global__ void el_er_kernel(const float* __restrict__ fsrc,
                             const float* __restrict__ attn_l,
                             const float* __restrict__ attn_r,
                             float* __restrict__ el, float* __restrict__ er, int N) {
    int wid  = (blockIdx.x * blockDim.x + threadIdx.x) >> 5;
    int lane = threadIdx.x & 31;
    if (wid >= N * HH) return;
    int n = wid >> 2, h = wid & 3;
    const float* row = fsrc + (size_t)n * GCOLS + h * DD;
    const float* al  = attn_l + h * DD;
    const float* ar  = attn_r + h * DD;
    float pl = 0.0f, pr = 0.0f;
    for (int d = lane; d < DD; d += 32) {
        float f = row[d];
        pl += f * al[d];
        pr += f * ar[d];
    }
    for (int off = 16; off; off >>= 1) {
        pl += __shfl_xor(pl, off, 32);
        pr += __shfl_xor(pr, off, 32);
    }
    if (lane == 0) { el[wid] = pl; er[wid] = pr; }
}

__global__ void edge_logits(const float* __restrict__ el, const float* __restrict__ er,
                            const int* __restrict__ src, const int* __restrict__ dst,
                            float* __restrict__ ebuf, float* __restrict__ mmax, int EH) {
    int i = blockIdx.x * blockDim.x + threadIdx.x;
    if (i >= EH) return;
    int e = i >> 2, h = i & 3;
    float v = el[src[e] * HH + h] + er[dst[e] * HH + h];
    v = (v > 0.0f) ? v : 0.2f * v;                 // leaky_relu 0.2
    ebuf[i] = v;
    atomicMaxF(&mmax[dst[e] * HH + h], v);
}

__global__ void edge_exp(const int* __restrict__ dst, float* __restrict__ ebuf,
                         const float* __restrict__ mmax, float* __restrict__ zsum, int EH) {
    int i = blockIdx.x * blockDim.x + threadIdx.x;
    if (i >= EH) return;
    int e = i >> 2, h = i & 3;
    float v = expf(ebuf[i] - mmax[dst[e] * HH + h]);
    ebuf[i] = v;
    atomicAdd(&zsum[dst[e] * HH + h], v);
}

// one block per edge: gat[dst] += alpha * fsrc[src]
__global__ void gat_scatter(const float* __restrict__ fsrc, const float* __restrict__ ebuf,
                            const float* __restrict__ zsum,
                            const int* __restrict__ src, const int* __restrict__ dst,
                            float* __restrict__ gat, int E) {
    int e = blockIdx.x;
    if (e >= E) return;
    int s = src[e], t = dst[e];
    for (int j = threadIdx.x; j < GCOLS; j += blockDim.x) {
        int h = j / DD;
        float alpha = ebuf[e * HH + h] / zsum[t * HH + h];
        atomicAdd(&gat[(size_t)t * GCOLS + j], alpha * fsrc[(size_t)s * GCOLS + j]);
    }
}

// ---------------- SAGE (gcn aggregator) ----------------

__global__ void sage_scatter(const float* __restrict__ feat,
                             const int* __restrict__ src1, const int* __restrict__ dst1,
                             float* __restrict__ neigh, float* __restrict__ deg, int E) {
    int e = blockIdx.x;
    if (e >= E) return;
    int s = src1[e], t = dst1[e];
    if (threadIdx.x == 0) atomicAdd(&deg[t], 1.0f);
    for (int d = threadIdx.x; d < DD; d += blockDim.x)
        atomicAdd(&neigh[(size_t)t * DD + d], feat[(size_t)s * DD + d]);
}

// h_neigh = (neigh + feat) / (deg + 1)  -> bf16 padded [N][KP]
__global__ void hneigh_kernel(const float* __restrict__ neigh, const float* __restrict__ feat,
                              const float* __restrict__ deg, unsigned short* __restrict__ hnbf,
                              int N) {
    int i = blockIdx.x * blockDim.x + threadIdx.x;
    int total = N * KP;
    if (i >= total) return;
    int n = i / KP, j = i - n * KP;
    float v = 0.0f;
    if (j < DD)
        v = (neigh[(size_t)n * DD + j] + feat[(size_t)n * DD + j]) / (deg[n] + 1.0f);
    hnbf[i] = f32_to_bf16(v);
}

// ---------------- final gated multi-head sum, one wave per node ----------------

__global__ void final_kernel(const float* __restrict__ feat, const float* __restrict__ gat,
                             const float* __restrict__ b_gat, const float* __restrict__ sage,
                             const float* __restrict__ b_sage, const float* __restrict__ Whw,
                             const float* __restrict__ bhw, float* __restrict__ out, int N) {
    int wid  = (blockIdx.x * blockDim.x + threadIdx.x) >> 5;
    int lane = threadIdx.x & 31;
    if (wid >= N) return;
    const float* grow = gat  + (size_t)wid * GCOLS;
    const float* srow = sage + (size_t)wid * DD;
    float w[HH + 1];
#pragma unroll
    for (int h = 0; h < HH + 1; ++h) {
        float p = 0.0f;
        if (h < HH) {
            for (int d = lane; d < DD; d += 32)
                p += (grow[h * DD + d] + b_gat[h * DD + d]) * Whw[d];
        } else {
            for (int d = lane; d < DD; d += 32)
                p += (srow[d] + b_sage[d]) * Whw[d];
        }
        for (int off = 16; off; off >>= 1) p += __shfl_xor(p, off, 32);
        p += bhw[0];
        w[h] = 1.0f / (1.0f + expf(-p));
    }
    for (int d = lane; d < DD; d += 32) {
        float acc = feat[(size_t)wid * DD + d];
#pragma unroll
        for (int h = 0; h < HH; ++h)
            acc += (grow[h * DD + d] + b_gat[h * DD + d]) * w[h];
        acc += (srow[d] + b_sage[d]) * w[HH];
        out[(size_t)wid * DD + d] = acc;
    }
}

// ---------------- host launcher ----------------

extern "C" void kernel_launch(void* const* d_in, const int* in_sizes, int n_in,
                              void* d_out, int out_size, void* d_ws, size_t ws_size,
                              hipStream_t stream) {
    const float* feat   = (const float*)d_in[0];
    const float* Wg     = (const float*)d_in[1];
    const float* attn_l = (const float*)d_in[2];
    const float* attn_r = (const float*)d_in[3];
    const float* b_gat  = (const float*)d_in[4];
    const float* Wsage  = (const float*)d_in[5];
    const float* b_sage = (const float*)d_in[6];
    const float* Whw    = (const float*)d_in[7];
    const float* bhw    = (const float*)d_in[8];
    const int*   src    = (const int*)d_in[9];
    const int*   dst    = (const int*)d_in[10];
    const int*   src1   = (const int*)d_in[11];
    const int*   dst1   = (const int*)d_in[12];
    float* out = (float*)d_out;

    int N  = in_sizes[0] / DD;
    int E  = in_sizes[9];
    int E1 = in_sizes[11];

    char* ws = (char*)d_ws;
    size_t off = 0;
    auto carve = [&](size_t bytes) -> void* {
        void* p = ws + off;
        off = (off + bytes + 255) & ~(size_t)255;
        return p;
    };
    unsigned short* featbf = (unsigned short*)carve((size_t)N * KP * 2);
    unsigned short* wgT    = (unsigned short*)carve((size_t)GPAD * KP * 2);
    unsigned short* wsT    = (unsigned short*)carve((size_t)SPAD * KP * 2);
    unsigned short* hnbf   = (unsigned short*)carve((size_t)N * KP * 2);
    float* fsrc  = (float*)carve((size_t)N * GCOLS * 4);
    float* gat   = (float*)carve((size_t)N * GCOLS * 4);
    float* el    = (float*)carve((size_t)N * HH * 4);
    float* er    = (float*)carve((size_t)N * HH * 4);
    float* mmax  = (float*)carve((size_t)N * HH * 4);
    float* zsum  = (float*)carve((size_t)N * HH * 4);
    float* ebuf  = (float*)carve((size_t)E * HH * 4);
    float* deg   = (float*)carve((size_t)N * 4);
    float* neigh = (float*)carve((size_t)N * DD * 4);
    float* sage  = (float*)carve((size_t)N * DD * 4);

    const int T = 256;
    auto gsz = [](int n, int t) { return (n + t - 1) / t; };

    // conversions
    conv_rows_bf16<<<gsz(N * KP, T), T, 0, stream>>>(feat, featbf, N, DD, KP);
    conv_bt_bf16<<<gsz(GPAD * KP, T), T, 0, stream>>>(Wg, wgT, DD, GCOLS, GPAD);
    conv_bt_bf16<<<gsz(SPAD * KP, T), T, 0, stream>>>(Wsage, wsT, DD, DD, SPAD);

    // inits
    fill_f32<<<gsz(N * GCOLS, T), T, 0, stream>>>(gat, 0.0f, N * GCOLS);
    fill_f32<<<gsz(N * HH, T), T, 0, stream>>>(zsum, 0.0f, N * HH);
    fill_f32<<<gsz(N * HH, T), T, 0, stream>>>(mmax, -INFINITY, N * HH);
    fill_f32<<<gsz(N, T), T, 0, stream>>>(deg, 0.0f, N);
    fill_f32<<<gsz(N * DD, T), T, 0, stream>>>(neigh, 0.0f, N * DD);

    // GEMM1: fsrc = feat @ Wg   [N][1200]
    {
        int mBlocks = (N + 31) / 32;
        int nBlocks = GPAD / 64;
        int waves = mBlocks * nBlocks;
        wmma_gemm_bf16_blk<<<gsz(waves * 32, 128), 128, 0, stream>>>(
            featbf, wgT, fsrc, N, mBlocks, nBlocks, GCOLS, GCOLS);
    }

    // attention logits + edge softmax
    el_er_kernel<<<gsz(N * HH * 32, T), T, 0, stream>>>(fsrc, attn_l, attn_r, el, er, N);
    edge_logits<<<gsz(E * HH, T), T, 0, stream>>>(el, er, src, dst, ebuf, mmax, E * HH);
    edge_exp<<<gsz(E * HH, T), T, 0, stream>>>(dst, ebuf, mmax, zsum, E * HH);
    gat_scatter<<<E, 128, 0, stream>>>(fsrc, ebuf, zsum, src, dst, gat, E);

    // SAGE gcn aggregation
    sage_scatter<<<E1, 128, 0, stream>>>(feat, src1, dst1, neigh, deg, E1);
    hneigh_kernel<<<gsz(N * KP, T), T, 0, stream>>>(neigh, feat, deg, hnbf, N);

    // GEMM2: sage = h_neigh @ Wsage  [N][300]
    {
        int mBlocks = (N + 31) / 32;
        int nBlocks = SPAD / 64;
        int waves = mBlocks * nBlocks;
        wmma_gemm_bf16_blk<<<gsz(waves * 32, 128), 128, 0, stream>>>(
            hnbf, wsT, sage, N, mBlocks, nBlocks, DD, DD);
    }

    // gated multi-head combine
    final_kernel<<<gsz(N * 32, T), T, 0, stream>>>(feat, gat, b_gat, sage, b_sage,
                                                   Whw, bhw, out, N);
}
